// Attention_53334903882008
// MI455X (gfx1250) — matrix operand order repaired and checked by reading, not compiled, not run on hardware
//
#include <hip/hip_runtime.h>

#define B_  8
#define T_  128
#define N_  64
#define D_  1024
#define H_  512
#define DP_ (D_ / 2)
#define XSTR 516   // uints per staged-X row: 2064B = 129*16B -> 16B-aligned frags,
                   // bank stride 516%64=4 words -> conflict-free across 16 rows
#define MAIN_THREADS 512

typedef __bf16 bf16x16 __attribute__((ext_vector_type(16)));
typedef float  floatx8 __attribute__((ext_vector_type(8)));
typedef float  floatx4 __attribute__((ext_vector_type(4)));

#if defined(__has_builtin)
#if __has_builtin(__builtin_amdgcn_cvt_pk_bf16_f32)
#define USE_CVT_PK_BF16 1
#endif
#endif
#ifndef USE_CVT_PK_BF16
#define USE_CVT_PK_BF16 0
#endif

__device__ __forceinline__ unsigned short f2bf(float f) {
  unsigned int x;
  __builtin_memcpy(&x, &f, 4);
  x += 0x7FFFu + ((x >> 16) & 1u);   // round-to-nearest-even
  return (unsigned short)(x >> 16);
}
__device__ __forceinline__ unsigned int pack2(float a, float b) {
#if USE_CVT_PK_BF16
  auto p = __builtin_amdgcn_cvt_pk_bf16_f32(a, b);  // v_cvt_pk_bf16_f32
  unsigned int u;
  __builtin_memcpy(&u, &p, 4);
  return u;
#else
  return (unsigned int)f2bf(a) | ((unsigned int)f2bf(b) << 16);
#endif
}
__device__ __forceinline__ float bf2f(unsigned int bits16) {
  unsigned int x = bits16 << 16;
  float f;
  __builtin_memcpy(&f, &x, 4);
  return f;
}

// ---- pack W_weight / U_weight [H][D] fp32 -> pair-packed bf16 [D/2][H] ----
__global__ __launch_bounds__(256) void pack_weight_kernel(
    const float* __restrict__ src, unsigned int* __restrict__ dst) {
  int idx = blockIdx.x * 256 + threadIdx.x;   // DP_*H_ = 262144 total
  int h  = idx & (H_ - 1);
  int dp = idx >> 9;
  const float* p = src + (size_t)h * D_ + 2 * dp;
  dst[idx] = pack2(p[0], p[1]);
}

// ---- pack hidden rows [512][1024] fp32 -> [512][512] packed bf16 pairs ----
__global__ __launch_bounds__(256) void pack_rows_kernel(
    const float* __restrict__ src, unsigned int* __restrict__ dst) {
  int idx = blockIdx.x * 256 + threadIdx.x;   // 131072 total, 4 floats each
  floatx4 f = reinterpret_cast<const floatx4*>(src)[idx];
  uint2 r;
  r.x = pack2(f.x, f.y);
  r.y = pack2(f.z, f.w);
  reinterpret_cast<uint2*>(dst)[idx] = r;
}

// ---- U_e[b][n][h] = hidden[n,b,:] . U_weight[h,:] + U_bias[h] ------------
__global__ __launch_bounds__(256) void ue_gemm_kernel(
    const unsigned int* __restrict__ Hp, const unsigned int* __restrict__ Up,
    const float* __restrict__ Ubias, float* __restrict__ Ue) {
  const int wave = threadIdx.x >> 5;
  const int lane = threadIdx.x & 31;
  const int tile = blockIdx.x * 8 + wave;   // 32x32 = 1024 tiles
  const int mt   = tile & 31;
  const int nt   = tile >> 5;
  const int hloc = lane & 15;
  const int half = lane >> 4;
  const int h    = nt * 16 + hloc;

  union Frag { bf16x16 v; unsigned int u[8]; uint4 q[2]; };
  floatx8 acc;
#pragma unroll
  for (int j = 0; j < 8; ++j) { acc[j] = 0.0f; }

  for (int k = 0; k < 32; ++k) {
    Frag afr, bfr;
    const int m = mt * 16 + hloc;
    const int abase = m * DP_ + k * 16 + half * 8;   // 16B-aligned
    afr.q[0] = *reinterpret_cast<const uint4*>(&Hp[abase]);
    afr.q[1] = *reinterpret_cast<const uint4*>(&Hp[abase + 4]);
    const int dpB = k * 16 + half * 8;
#pragma unroll
    for (int v = 0; v < 8; ++v) {
      bfr.u[v] = Up[(dpB + v) * H_ + h];
    }
    acc = __builtin_amdgcn_wmma_f32_16x16x32_bf16(
        false, afr.v, false, bfr.v, (short)0, acc, false, false);
  }

  const float ub = Ubias[h];
#pragma unroll
  for (int r = 0; r < 8; ++r) {
    const int rr = mt * 16 + r + half * 8;  // row = n*B + b
    const int n  = rr >> 3;
    const int bb = rr & 7;
    Ue[((size_t)bb * N_ + n) * H_ + h] = acc[r] + ub;
  }
}

// ---- fused attention: one workgroup per (b,t), 16 waves ------------------
__global__ __launch_bounds__(MAIN_THREADS) void attn_main_kernel(
    const float* __restrict__ img, const unsigned int* __restrict__ Wp,
    const float* __restrict__ Wbias, const float* __restrict__ wvec,
    const float* __restrict__ Ue, float* __restrict__ out) {
  __shared__ unsigned int sX[N_ * XSTR];   // 132 KB: img tile as bf16 pairs
  __shared__ float sEpart[32 * N_];        // per-ntile partial e (deterministic)
  __shared__ float sE[N_];
  __shared__ float sAlpha[N_];
  __shared__ float sWb[H_];
  __shared__ float sWv[H_];

  const int bt  = blockIdx.x;              // 0..1023
  const int b   = bt >> 7;
  const int tid = threadIdx.x;

  const float* __restrict__ Xg = img + (size_t)bt * (N_ * D_);

  // Stage img[b,t] (64x1024 fp32) into LDS as packed bf16 pairs.
  // Non-temporal: img is streamed exactly once -> don't evict hot L2 (Wp/Ue).
  for (int i = tid; i < (N_ * D_) / 4; i += MAIN_THREADS) {
    floatx4 f = __builtin_nontemporal_load(reinterpret_cast<const floatx4*>(Xg) + i);
    const int e0 = i * 4;
    const int n  = e0 >> 10;
    const int d0 = e0 & (D_ - 1);
    uint2 r;
    r.x = pack2(f.x, f.y);
    r.y = pack2(f.z, f.w);
    *reinterpret_cast<uint2*>(&sX[n * XSTR + (d0 >> 1)]) = r;
  }
  for (int i = tid; i < H_; i += MAIN_THREADS) { sWb[i] = Wbias[i]; sWv[i] = wvec[i]; }
  __syncthreads();

  const int wave = tid >> 5;   // 0..15
  const int lane = tid & 31;
  const int hloc = lane & 15;
  const int half = lane >> 4;
  const float* __restrict__ UeB = Ue + (size_t)b * (N_ * H_);

  union Frag { bf16x16 v; unsigned int u[8]; uint4 q[2]; };

  for (int nti = 0; nti < 2; ++nti) {
    const int nt = wave + nti * 16;  // each of 32 n-tiles handled by one wave
    const int h  = nt * 16 + hloc;

    floatx8 acc[4];
#pragma unroll
    for (int mt = 0; mt < 4; ++mt) {
#pragma unroll
      for (int j = 0; j < 8; ++j) { acc[mt][j] = 0.0f; }
    }

    for (int k = 0; k < 32; ++k) {
      Frag bfr;  // B fragment: reused across 4 m-tiles (cuts W L2 traffic 4x)
      const int dpB = k * 16 + half * 8;
#pragma unroll
      for (int v = 0; v < 8; ++v) {
        bfr.u[v] = Wp[(dpB + v) * H_ + h];
      }
#pragma unroll
      for (int mt = 0; mt < 4; ++mt) {
        Frag afr;
        const int m    = mt * 16 + hloc;
        const int base = m * XSTR + k * 16 + half * 8;   // 16B-aligned
        afr.q[0] = *reinterpret_cast<const uint4*>(&sX[base]);      // ds_load_b128
        afr.q[1] = *reinterpret_cast<const uint4*>(&sX[base + 4]);  // ds_load_b128
        acc[mt] = __builtin_amdgcn_wmma_f32_16x16x32_bf16(
            false, afr.v, false, bfr.v, (short)0, acc[mt], false, false);
      }
    }

    // epilogue: e[n] += sum_h w[h] * tanh(We + Wbias + Ue)
    const float wbh = sWb[h];
    const float wvh = sWv[h];
#pragma unroll
    for (int mt = 0; mt < 4; ++mt) {
#pragma unroll
      for (int r = 0; r < 8; ++r) {
        const int n = mt * 16 + r + half * 8;
        float v = acc[mt][r] + wbh + UeB[n * H_ + h];
        float c = wvh * tanhf(v);
        c += __shfl_xor(c, 1);   // butterfly within each 16-lane half
        c += __shfl_xor(c, 2);
        c += __shfl_xor(c, 4);
        c += __shfl_xor(c, 8);
        if (hloc == 0) { sEpart[nt * N_ + n] = c; }
      }
    }
  }
  __syncthreads();

  // deterministic reduce of the 32 per-ntile partials, then softmax over N.
  if (tid < N_) {
    float e = 0.0f;
    for (int nt = 0; nt < 32; ++nt) { e += sEpart[nt * N_ + tid]; }
    sE[tid] = e;   // w_bias omitted: constant shift cancels in softmax
  }
  __syncthreads();
  if (tid == 0) {
    float mx = sE[0];
    for (int n = 1; n < N_; ++n) { mx = fmaxf(mx, sE[n]); }
    float s = 0.0f;
    for (int n = 0; n < N_; ++n) {
      float e = __expf(sE[n] - mx);
      sAlpha[n] = e;
      s += e;
    }
    const float inv = 1.0f / s;
    for (int n = 0; n < N_; ++n) { sAlpha[n] *= inv; }
  }
  __syncthreads();

  // phi[b,t,d] = sum_n alpha[n] * img[b,t,n,d]  (from staged LDS tile)
  float* __restrict__ op = out + (size_t)bt * D_;
  for (int dp = tid; dp < DP_; dp += MAIN_THREADS) {
    float a0 = 0.0f, a1 = 0.0f;
    for (int n = 0; n < N_; ++n) {
      const unsigned int u = sX[n * XSTR + dp];
      const float al = sAlpha[n];
      a0 += al * bf2f(u & 0xFFFFu);
      a1 += al * bf2f(u >> 16);
    }
    __builtin_nontemporal_store(a0, op + 2 * dp);
    __builtin_nontemporal_store(a1, op + 2 * dp + 1);
  }
}

extern "C" void kernel_launch(void* const* d_in, const int* in_sizes, int n_in,
                              void* d_out, int out_size, void* d_ws, size_t ws_size,
                              hipStream_t stream) {
  (void)in_sizes; (void)n_in; (void)out_size; (void)ws_size;
  const float* img = (const float*)d_in[0];   // [B,T,N,D]
  const float* hid = (const float*)d_in[1];   // [N,B,D]
  const float* Wwt = (const float*)d_in[2];   // [H,D]
  const float* Wb  = (const float*)d_in[3];   // [H]
  const float* Uwt = (const float*)d_in[4];   // [H,D]
  const float* Ub  = (const float*)d_in[5];   // [H]
  const float* wv  = (const float*)d_in[6];   // [1,H]
  // d_in[7] = w_bias: constant pre-softmax shift, mathematically cancels.
  float* out = (float*)d_out;                 // [B,T,D]

  // workspace layout (4 MB total): Wp | Up | Hp | Ue, 1 MB each
  char* ws = (char*)d_ws;
  unsigned int* Wp = (unsigned int*)(ws + (size_t)0 * (1u << 20));
  unsigned int* Up = (unsigned int*)(ws + (size_t)1 * (1u << 20));
  unsigned int* Hp = (unsigned int*)(ws + (size_t)2 * (1u << 20));
  float*        Ue = (float*)       (ws + (size_t)3 * (1u << 20));

  pack_weight_kernel<<<(DP_ * H_) / 256, 256, 0, stream>>>(Wwt, Wp);
  pack_weight_kernel<<<(DP_ * H_) / 256, 256, 0, stream>>>(Uwt, Up);
  pack_rows_kernel  <<<(N_ * B_ * DP_ / 2) / 256, 256, 0, stream>>>(hid, Hp);
  ue_gemm_kernel    <<<128, 256, 0, stream>>>(Hp, Up, Ub, Ue);
  attn_main_kernel  <<<B_ * T_, MAIN_THREADS, 0, stream>>>(img, Wp, Wb, wv, Ue, out);
}
